// content_enc_33887291966171
// MI455X (gfx1250) — matrix operand order repaired and checked by reading, not compiled, not run on hardware
//
#include <hip/hip_runtime.h>
#include <hip/hip_bf16.h>

typedef __attribute__((ext_vector_type(16))) _Float16 v16h;
typedef __attribute__((ext_vector_type(2)))  _Float16 hpair;
typedef __attribute__((ext_vector_type(8)))  float    v8f;

#define DCH   1024
#define LOUT  511
#define NB    32
#define K1    4096
#define KCODE 2048
#define BN_EPS 1e-5f
#define NL    (NB*LOUT)          // 16352 rows (b,l)
#define QSZ   (NB*DCH*LOUT)      // 16744448

// Fragment-order layouts (k order chosen by this code, wave32):
//  A frag (16x32 f16): lane(ml,half) reads k = half*16 + j, j=0..15
//  B frag (32x16 f16): lane(col=ml,half) reads k = half*16 + j
// Weights are pre-swizzled to fragment order in GLOBAL memory (L2-resident),
// so a lane's A fragment is one aligned 32B read (2x global_load_b128).
// The B tile is staged in LDS in fragment order (2x ds_load_b128 per frag)
// with a ping-pong buffer -> one barrier per k-step.

#define WMMA(a,b,c) __builtin_amdgcn_wmma_f32_16x16x32_f16(false,(a),false,(b),(short)0,(c),false,false)

// ---------------------------------------------------------------------------
// Prep: swizzle f32 weights [1024 x Kdim] into f16 A-fragment order.
// wsz[(((g*KC + kc)*2 + halfk)*16 + ml)*16 + j], Kdim = 1<<kshift.
// ---------------------------------------------------------------------------
__global__ __launch_bounds__(256) void w_swizzle(const float* __restrict__ w,
                                                 _Float16* __restrict__ wsz,
                                                 int kshift)
{
    int idx = blockIdx.x * 256 + threadIdx.x;
    int o = idx >> kshift, k = idx & ((1 << kshift) - 1);
    int KC = 1 << (kshift - 5);
    int g = o >> 4, ml = o & 15;
    int kc = k >> 5, k32 = k & 31;
    int hf = (k32 >> 4) & 1, j = k32 & 15;
    wsz[(((g * KC + kc) * 2 + hf) * 16 + ml) * 16 + j] = (_Float16)w[idx];
}

// ---------------------------------------------------------------------------
// Prep: fold BN into scale/shift: y = x*scale + shift
// ---------------------------------------------------------------------------
__global__ __launch_bounds__(256) void bn_prep(const float* __restrict__ meanv,
                                               const float* __restrict__ varv,
                                               const float* __restrict__ g,
                                               const float* __restrict__ be,
                                               float* __restrict__ scale,
                                               float* __restrict__ shift)
{
    int c = blockIdx.x * 256 + threadIdx.x;
    float s = g[c] * rsqrtf(varv[c] + BN_EPS);
    scale[c] = s;
    shift[c] = be[c] - meanv[c] * s;
}

// ---------------------------------------------------------------------------
// Kernel 1: conv1 (k=4, stride=2) as im2col GEMM. Block tile 128(O) x 32(L);
// 8 waves; A fragments direct from L2-resident swizzled weights; B tile in
// ping-pong LDS (one barrier per k-step); 2 WMMAs per wave per k-step.
// ---------------------------------------------------------------------------
__global__ __launch_bounds__(256) void conv1_gemm(const float* __restrict__ x,
                                                  const _Float16* __restrict__ w1s,
                                                  const float* __restrict__ b1,
                                                  float* __restrict__ h1)
{
    __shared__ __align__(32) _Float16 Bswz[2][2 * 32 * 16];
    const int n  = blockIdx.z;
    const int o0 = blockIdx.y * 128;
    const int l0 = blockIdx.x * 32;
    const int tid  = threadIdx.x;
    const int lane = tid & 31, wave = tid >> 5;
    const int half = lane >> 4, ml = lane & 15;
    const int KC = K1 / 32;
    const _Float16* abase = w1s + (((blockIdx.y * 8 + wave) * KC) * 2 + half) * 256 + ml * 16;

    // Per-thread fixed B-loader geometry: 2 (k-pair, col) elements
    int  sadr[2], xoff[2];
    bool ok[2];
    #pragma unroll
    for (int i = 0; i < 2; ++i) {
        int e = tid + i * 256;
        int kk = (e >> 5) * 2, ll = e & 31;
        int lg = l0 + ll;
        ok[i]   = (lg < LOUT);
        xoff[i] = (n * DCH + (kk >> 2)) * 1024 + 2 * lg + (kk & 3); // taps contiguous
        sadr[i] = ((kk >> 4) * 32 + ll) * 16 + (kk & 15);
    }

    v8f acc0 = {}, acc1 = {};
    // Prologue: stage chunk 0
    #pragma unroll
    for (int i = 0; i < 2; ++i) {
        float a0 = 0.f, a1 = 0.f;
        if (ok[i]) { const float* q = x + xoff[i]; a0 = q[0]; a1 = q[1]; }
        hpair pk; pk.x = (_Float16)a0; pk.y = (_Float16)a1;
        *(hpair*)(&Bswz[0][sadr[i]]) = pk;
    }
    __syncthreads();

    int p = 0;
    for (int kc = 0; kc < KC; ++kc) {
        hpair nxt[2];
        if (kc + 1 < KC) {
            #pragma unroll
            for (int i = 0; i < 2; ++i) {
                float a0 = 0.f, a1 = 0.f;
                if (ok[i]) { const float* q = x + xoff[i] + (kc + 1) * 8192; a0 = q[0]; a1 = q[1]; }
                nxt[i].x = (_Float16)a0; nxt[i].y = (_Float16)a1;
            }
        }
        v16h a  = *(const v16h*)(abase + (size_t)kc * 512);
        v16h bA = *(const v16h*)(&Bswz[p][(half * 32 + ml) * 16]);
        v16h bB = *(const v16h*)(&Bswz[p][(half * 32 + 16 + ml) * 16]);
        acc0 = WMMA(a, bA, acc0);
        acc1 = WMMA(a, bB, acc1);
        if (kc + 1 < KC) {
            #pragma unroll
            for (int i = 0; i < 2; ++i) *(hpair*)(&Bswz[p ^ 1][sadr[i]]) = nxt[i];
        }
        __syncthreads();
        p ^= 1;
    }

    int lgA = l0 + ml, lgB = l0 + 16 + ml;
    if (lgA < LOUT) {
        #pragma unroll
        for (int r = 0; r < 8; ++r) {
            int og = o0 + wave * 16 + r + 8 * half;
            h1[(n * DCH + og) * LOUT + lgA] = acc0[r] + b1[og];
        }
    }
    if (lgB < LOUT) {
        #pragma unroll
        for (int r = 0; r < 8; ++r) {
            int og = o0 + wave * 16 + r + 8 * half;
            h1[(n * DCH + og) * LOUT + lgB] = acc1[r] + b1[og];
        }
    }
}

// ---------------------------------------------------------------------------
// Kernel 2: BN stats (biased var over (N, L) per channel)
// ---------------------------------------------------------------------------
__global__ __launch_bounds__(256) void bn_stats(const float* __restrict__ h,
                                                float* __restrict__ meanv,
                                                float* __restrict__ varv)
{
    __shared__ float s1[256], s2[256];
    const int c = blockIdx.x;
    float a = 0.f, b = 0.f;
    for (int i = threadIdx.x; i < NL; i += 256) {
        int n = i / LOUT, l = i - n * LOUT;
        float v = h[(n * DCH + c) * LOUT + l];
        a += v; b += v * v;
    }
    s1[threadIdx.x] = a; s2[threadIdx.x] = b;
    __syncthreads();
    for (int st = 128; st > 0; st >>= 1) {
        if (threadIdx.x < st) {
            s1[threadIdx.x] += s1[threadIdx.x + st];
            s2[threadIdx.x] += s2[threadIdx.x + st];
        }
        __syncthreads();
    }
    if (threadIdx.x == 0) {
        float m = s1[0] / (float)NL;
        meanv[c] = m;
        varv[c]  = s2[0] / (float)NL - m * m;
    }
}

// ---------------------------------------------------------------------------
// Kernel 3: r = conv1x1(relu(bn1(h1))); h2 = h1 + r + b2  (fused epilogue)
// ---------------------------------------------------------------------------
__global__ __launch_bounds__(256) void conv2_gemm(const float* __restrict__ h1,
                                                  const _Float16* __restrict__ w2s,
                                                  const float* __restrict__ b2,
                                                  const float* __restrict__ scale1,
                                                  const float* __restrict__ shift1,
                                                  float* __restrict__ h2)
{
    __shared__ __align__(32) _Float16 Bswz[2][2 * 32 * 16];
    const int n  = blockIdx.z;
    const int o0 = blockIdx.y * 128;
    const int l0 = blockIdx.x * 32;
    const int tid  = threadIdx.x;
    const int lane = tid & 31, wave = tid >> 5;
    const int half = lane >> 4, ml = lane & 15;
    const int KC = DCH / 32;
    const _Float16* abase = w2s + (((blockIdx.y * 8 + wave) * KC) * 2 + half) * 256 + ml * 16;

    int  sadr[2], kk0[2], hoff[2];
    bool ok[2];
    #pragma unroll
    for (int i = 0; i < 2; ++i) {
        int e = tid + i * 256;
        int kk = (e >> 5) * 2, ll = e & 31;
        int lg = l0 + ll;
        ok[i]   = (lg < LOUT);
        kk0[i]  = kk;
        hoff[i] = (n * DCH + kk) * LOUT + lg;
        sadr[i] = ((kk >> 4) * 32 + ll) * 16 + (kk & 15);
    }

    v8f acc0 = {}, acc1 = {};
    #pragma unroll
    for (int i = 0; i < 2; ++i) {
        float a0 = 0.f, a1 = 0.f;
        if (ok[i]) {
            int c0 = kk0[i];
            a0 = fmaxf(h1[hoff[i]]        * scale1[c0]     + shift1[c0],     0.f);
            a1 = fmaxf(h1[hoff[i] + LOUT] * scale1[c0 + 1] + shift1[c0 + 1], 0.f);
        }
        hpair pk; pk.x = (_Float16)a0; pk.y = (_Float16)a1;
        *(hpair*)(&Bswz[0][sadr[i]]) = pk;
    }
    __syncthreads();

    int p = 0;
    for (int kc = 0; kc < KC; ++kc) {
        hpair nxt[2];
        if (kc + 1 < KC) {
            #pragma unroll
            for (int i = 0; i < 2; ++i) {
                float a0 = 0.f, a1 = 0.f;
                if (ok[i]) {
                    int c0 = (kc + 1) * 32 + kk0[i];
                    int ho = hoff[i] + (kc + 1) * 32 * LOUT;
                    a0 = fmaxf(h1[ho]        * scale1[c0]     + shift1[c0],     0.f);
                    a1 = fmaxf(h1[ho + LOUT] * scale1[c0 + 1] + shift1[c0 + 1], 0.f);
                }
                nxt[i].x = (_Float16)a0; nxt[i].y = (_Float16)a1;
            }
        }
        v16h a  = *(const v16h*)(abase + (size_t)kc * 512);
        v16h bA = *(const v16h*)(&Bswz[p][(half * 32 + ml) * 16]);
        v16h bB = *(const v16h*)(&Bswz[p][(half * 32 + 16 + ml) * 16]);
        acc0 = WMMA(a, bA, acc0);
        acc1 = WMMA(a, bB, acc1);
        if (kc + 1 < KC) {
            #pragma unroll
            for (int i = 0; i < 2; ++i) *(hpair*)(&Bswz[p ^ 1][sadr[i]]) = nxt[i];
        }
        __syncthreads();
        p ^= 1;
    }

    int lgA = l0 + ml, lgB = l0 + 16 + ml;
    if (lgA < LOUT) {
        #pragma unroll
        for (int r = 0; r < 8; ++r) {
            int og = o0 + wave * 16 + r + 8 * half;
            h2[(n * DCH + og) * LOUT + lgA] =
                acc0[r] + h1[(n * DCH + og) * LOUT + lgA] + b2[og];
        }
    }
    if (lgB < LOUT) {
        #pragma unroll
        for (int r = 0; r < 8; ++r) {
            int og = o0 + wave * 16 + r + 8 * half;
            h2[(n * DCH + og) * LOUT + lgB] =
                acc1[r] + h1[(n * DCH + og) * LOUT + lgB] + b2[og];
        }
    }
}

// ---------------------------------------------------------------------------
// Kernel 4a: codebook -> f16 copy
// ---------------------------------------------------------------------------
__global__ __launch_bounds__(256) void cb_to_f16(const float* __restrict__ cb,
                                                 _Float16* __restrict__ cbh)
{
    int i = (blockIdx.x * 256 + threadIdx.x) * 4;
    #pragma unroll
    for (int j = 0; j < 4; ++j) cbh[i + j] = (_Float16)cb[i + j];
}

// ---------------------------------------------------------------------------
// Kernel 4b: codebook row norms e2[k] = sum_d cb[k,d]^2
// ---------------------------------------------------------------------------
__global__ __launch_bounds__(256) void e2_init(const float* __restrict__ cb,
                                               float* __restrict__ e2)
{
    __shared__ float s[256];
    const int k = blockIdx.x;
    float a = 0.f;
    for (int d = threadIdx.x; d < DCH; d += 256) {
        float v = cb[k * DCH + d];
        a += v * v;
    }
    s[threadIdx.x] = a;
    __syncthreads();
    for (int st = 128; st > 0; st >>= 1) {
        if (threadIdx.x < st) s[threadIdx.x] += s[threadIdx.x + st];
        __syncthreads();
    }
    if (threadIdx.x == 0) e2[k] = s[0];
}

// ---------------------------------------------------------------------------
// Kernel 5: VQ argmin. 16 z-rows staged fragment-swizzled in LDS; 8 waves
// sweep the 128 codebook tiles via WMMA, tracking running (min, argmin) of
// e2[k] - 2*z.e; cross-wave LDS reduce with first-index tie-breaking.
// ---------------------------------------------------------------------------
__global__ __launch_bounds__(256) void vq_argmin(const float* __restrict__ h2,
                                                 const float* __restrict__ scale2,
                                                 const float* __restrict__ shift2,
                                                 const _Float16* __restrict__ cbh,
                                                 const float* __restrict__ e2,
                                                 int*   __restrict__ ids,
                                                 float* __restrict__ ids_f)
{
    __shared__ __align__(32) _Float16 zswz[32 * 512];   // [chunk*2+half][row][j]
    __shared__ float rv[8 * 16 * 16];
    __shared__ int   ri[8 * 16 * 16];
    const int tid  = threadIdx.x;
    const int lane = tid & 31, wave = tid >> 5;
    const int half = lane >> 4, ml = lane & 15;
    const int m0 = blockIdx.x * 16;

    for (int e = tid; e < 16 * 512; e += 256) {
        int row = e >> 9, d = (e & 511) * 2;
        int m = m0 + row, bb = m / LOUT, l = m - bb * LOUT;
        float z0 = h2[(bb * DCH + d)     * LOUT + l] * scale2[d]     + shift2[d];
        float z1 = h2[(bb * DCH + d + 1) * LOUT + l] * scale2[d + 1] + shift2[d + 1];
        int k32 = d & 31, chunk = d >> 5;
        int addr = ((chunk * 2 + ((k32 >> 4) & 1)) * 16 + row) * 16
                 + (k32 & 15);
        hpair pk; pk.x = (_Float16)z0; pk.y = (_Float16)z1;
        *(hpair*)(zswz + addr) = pk;
    }
    __syncthreads();

    float minv[8]; int mini[8];
    #pragma unroll
    for (int r = 0; r < 8; ++r) { minv[r] = 3.4e38f; mini[r] = 0; }

    for (int nt = wave; nt < KCODE / 16; nt += 8) {
        v8f acc = {};
        const int code = nt * 16 + ml;
        for (int kcc = 0; kcc < 32; ++kcc) {
            v16h a = *(const v16h*)(zswz + ((kcc * 2 + half) * 16 + ml) * 16);
            v16h b = *(const v16h*)(cbh + code * DCH + kcc * 32 + half * 16);
            acc = WMMA(a, b, acc);
        }
        float ek = e2[code];
        #pragma unroll
        for (int r = 0; r < 8; ++r) {
            float dv = ek - 2.0f * acc[r];
            if (dv < minv[r]) { minv[r] = dv; mini[r] = code; }
        }
    }
    #pragma unroll
    for (int r = 0; r < 8; ++r) {
        int row16 = r + 8 * half;
        rv[(wave * 16 + row16) * 16 + ml] = minv[r];
        ri[(wave * 16 + row16) * 16 + ml] = mini[r];
    }
    __syncthreads();
    if (tid < 16) {
        float best = 3.4e38f; int bi = 0x7fffffff;
        for (int w = 0; w < 8; ++w)
            for (int c2 = 0; c2 < 16; ++c2) {
                float v = rv[(w * 16 + tid) * 16 + c2];
                int  iv = ri[(w * 16 + tid) * 16 + c2];
                if (v < best || (v == best && iv < bi)) { best = v; bi = iv; }
            }
        int m = m0 + tid;
        ids[m]   = bi;
        ids_f[m] = (float)bi;
    }
}

// ---------------------------------------------------------------------------
// Kernel 6: q_st = codebook[id]; per-row partial sum of (z-q)^2 in double.
// ---------------------------------------------------------------------------
__global__ __launch_bounds__(256) void qst_loss(const float* __restrict__ h2,
                                                const float* __restrict__ scale2,
                                                const float* __restrict__ shift2,
                                                const float* __restrict__ cb,
                                                const int* __restrict__ ids,
                                                float* __restrict__ outq,
                                                double* __restrict__ lsum)
{
    __shared__ double sd[256];
    const int m  = blockIdx.x;
    const int bb = m / LOUT, l = m - bb * LOUT;
    const int id = ids[m];
    double s = 0.0;
    for (int d = threadIdx.x; d < DCH; d += 256) {
        float q = cb[id * DCH + d];
        float z = h2[(bb * DCH + d) * LOUT + l] * scale2[d] + shift2[d];
        outq[(bb * DCH + d) * LOUT + l] = q;
        float diff = z - q;
        s += (double)diff * (double)diff;
    }
    sd[threadIdx.x] = s;
    __syncthreads();
    for (int st = 128; st > 0; st >>= 1) {
        if (threadIdx.x < st) sd[threadIdx.x] += sd[threadIdx.x + st];
        __syncthreads();
    }
    if (threadIdx.x == 0) lsum[m] = sd[0];
}

// ---------------------------------------------------------------------------
// Kernel 7: reduce loss partials; commitment == codebook_loss numerically.
// ---------------------------------------------------------------------------
__global__ __launch_bounds__(256) void finalize(const double* __restrict__ lsum,
                                                float* __restrict__ outl)
{
    __shared__ double sd[256];
    double s = 0.0;
    for (int i = threadIdx.x; i < NL; i += 256) s += lsum[i];
    sd[threadIdx.x] = s;
    __syncthreads();
    for (int st = 128; st > 0; st >>= 1) {
        if (threadIdx.x < st) sd[threadIdx.x] += sd[threadIdx.x + st];
        __syncthreads();
    }
    if (threadIdx.x == 0) {
        float c = (float)(sd[0] / (double)((long long)NB * LOUT * DCH));
        outl[0] = c;   // commitment
        outl[1] = c;   // codebook_loss (stop_gradient doesn't change the value)
    }
}

// ---------------------------------------------------------------------------
extern "C" void kernel_launch(void* const* d_in, const int* in_sizes, int n_in,
                              void* d_out, int out_size, void* d_ws, size_t ws_size,
                              hipStream_t stream)
{
    const float* x   = (const float*)d_in[0];
    const float* w1  = (const float*)d_in[1];
    const float* b1  = (const float*)d_in[2];
    const float* g1  = (const float*)d_in[3];
    const float* be1 = (const float*)d_in[4];
    const float* w2  = (const float*)d_in[5];
    const float* b2  = (const float*)d_in[6];
    const float* g2  = (const float*)d_in[7];
    const float* be2 = (const float*)d_in[8];
    const float* cb  = (const float*)d_in[9];

    char* ws = (char*)d_ws;
    float*    h1    = (float*)(ws);                       // 66,977,792 B
    float*    h2    = (float*)(ws + 66977792);            // 66,977,792 B
    float*    mean1 = (float*)(ws + 133955584);           // 4 x 1024 f32
    float*    var1  = mean1 + 1024;
    float*    mean2 = mean1 + 2048;
    float*    var2  = mean1 + 3072;
    float*    e2    = (float*)(ws + 133971968);           // 2048 f32
    int*      ids   = (int*)  (ws + 133980160);           // 16352 i32
    double*   lsum  = (double*)(ws + 134045568);          // 16352 f64 (8B aligned)
    _Float16* cbh   = (_Float16*)(ws + 134176384);        // 2048*1024 f16 (32B aligned)
    _Float16* w1s   = (_Float16*)(ws + 138370688);        // 1024*4096 f16 swizzled
    _Float16* w2s   = (_Float16*)(ws + 146759296);        // 1024*1024 f16 swizzled
    float*    scale1= (float*)(ws + 148856448);           // 4 x 1024 f32
    float*    shift1= scale1 + 1024;
    float*    scale2= scale1 + 2048;
    float*    shift2= scale1 + 3072;

    float* outq  = (float*)d_out;           // [32, 1024, 511]
    float* idsf  = outq + QSZ;              // [32, 511]
    float* outl  = idsf + NL;               // commitment, codebook_loss

    w_swizzle <<<(1024 * K1) / 256, 256, 0, stream>>>(w1, w1s, 12);
    w_swizzle <<<(1024 * DCH) / 256, 256, 0, stream>>>(w2, w2s, 10);
    cb_to_f16 <<<(KCODE * DCH) / 1024, 256, 0, stream>>>(cb, cbh);
    e2_init   <<<KCODE, 256, 0, stream>>>(cb, e2);

    conv1_gemm<<<dim3(16, 8, 32), 256, 0, stream>>>(x, w1s, b1, h1);
    bn_stats  <<<1024, 256, 0, stream>>>(h1, mean1, var1);
    bn_prep   <<<4, 256, 0, stream>>>(mean1, var1, g1, be1, scale1, shift1);
    conv2_gemm<<<dim3(16, 8, 32), 256, 0, stream>>>(h1, w2s, b2, scale1, shift1, h2);
    bn_stats  <<<1024, 256, 0, stream>>>(h2, mean2, var2);
    bn_prep   <<<4, 256, 0, stream>>>(mean2, var2, g2, be2, scale2, shift2);

    vq_argmin <<<NL / 16, 256, 0, stream>>>(h2, scale2, shift2, cbh, e2, ids, idsf);
    qst_loss  <<<NL, 256, 0, stream>>>(h2, scale2, shift2, cb, ids, outq, lsum);
    finalize  <<<1, 256, 0, stream>>>(lsum, outl);
}
